// SD21_67886253080730
// MI455X (gfx1250) — matrix operand order repaired and checked by reading, not compiled, not run on hardware
//
#include <hip/hip_runtime.h>
#include <stdint.h>

// ---------------- problem constants (from reference) ----------------
#define NN   65536    // nodes per drug
#define NE   131072   // edges per drug
#define NGR  1024     // graphs per batch
#define NPG  64       // nodes per graph (contiguous)
#define EPG  128      // edges per graph (contiguous)
#define NH   4        // attention heads
#define HC   256      // NH * co
#define CD   64       // node channels (constant through net)
#define HIDD 128      // HID
#define NCLS 65

typedef __attribute__((ext_vector_type(16))) __bf16 bf16x16;
typedef __attribute__((ext_vector_type(8)))  __bf16 bf16x8;
typedef __attribute__((ext_vector_type(8)))  float  f32x8;

__device__ __forceinline__ float eluf(float v, int doElu) {
  return doElu ? (v > 0.f ? v : (__expf(v) - 1.f)) : v;
}
__device__ __forceinline__ unsigned f2key(float f) {
  unsigned u = __float_as_uint(f);
  return (u & 0x80000000u) ? ~u : (u | 0x80000000u);
}
__device__ __forceinline__ float key2f(unsigned u) {
  return (u & 0x80000000u) ? __uint_as_float(u ^ 0x80000000u) : __uint_as_float(~u);
}

// ================= bf16 WMMA GEMM: C = act(A[M,K] @ W[K,N] + bias) =================
// REQUIRES (true for every call in this net): M % 128 == 0, K % 16 == 0, Nc % 16 == 0.
#define BM 128
#define BN 64
#define BK 32
#define LDSP 40   // padded LDS row stride (elements); multiple of 8 keeps 16B alignment

__global__ __launch_bounds__(256) void gemm_wmma_kernel(
    const float* __restrict__ A, const float* __restrict__ W,
    const float* __restrict__ bias, float* __restrict__ C,
    int M, int K, int Nc, int act)
{
  __shared__ __bf16 sA[BM * LDSP];
  __shared__ __bf16 sB[BN * LDSP];
  const int tid  = threadIdx.x;
  const int wave = tid >> 5;
  const int lane = tid & 31;
  const int rowBase = blockIdx.y * BM;
  const int colBase = blockIdx.x * BN;

  f32x8 acc[4];
#pragma unroll
  for (int j = 0; j < 4; ++j)
#pragma unroll
    for (int t = 0; t < 8; ++t) acc[j][t] = 0.f;

  // ---- staging: vectorized loads (float4 / b128), per-vector K guard only ----
  float4 aR[4], bR[2];
  auto loadA = [&](int k0) {
#pragma unroll
    for (int u = 0; u < 4; ++u) {
      int v = u * 256 + tid;               // over BM*BK/4 = 1024 float4s
      int row = v >> 3, c4 = (v & 7) << 2;
      int gc = k0 + c4;
      float4 z; z.x = z.y = z.z = z.w = 0.f;
      aR[u] = z;
      if (gc < K)
        aR[u] = *(const float4*)(A + (size_t)(rowBase + row) * K + gc);
    }
  };
  auto loadB = [&](int k0) {
#pragma unroll
    for (int u = 0; u < 2; ++u) {
      int v = u * 256 + tid;               // over BN*BK/4 = 512 float4s
      int k = v >> 4, n4 = (v & 15) << 2;
      int gc = k0 + k, gn = colBase + n4;
      float4 z; z.x = z.y = z.z = z.w = 0.f;
      bR[u] = z;
      if (gc < K && gn < Nc)
        bR[u] = *(const float4*)(W + (size_t)gc * Nc + gn);
    }
  };
  auto storeTiles = [&]() {
#pragma unroll
    for (int u = 0; u < 4; ++u) {
      int v = u * 256 + tid;
      int row = v >> 3, c4 = (v & 7) << 2;
      __bf16* p = &sA[row * LDSP + c4];    // 8B-aligned, contiguous -> ds_store_b64
      p[0] = (__bf16)aR[u].x; p[1] = (__bf16)aR[u].y;
      p[2] = (__bf16)aR[u].z; p[3] = (__bf16)aR[u].w;
    }
#pragma unroll
    for (int u = 0; u < 2; ++u) {
      int v = u * 256 + tid;
      int k = v >> 4, n4 = (v & 15) << 2;  // transposed scatter: sB[n][k] = W[k][n]
      sB[(n4 + 0) * LDSP + k] = (__bf16)bR[u].x;
      sB[(n4 + 1) * LDSP + k] = (__bf16)bR[u].y;
      sB[(n4 + 2) * LDSP + k] = (__bf16)bR[u].z;
      sB[(n4 + 3) * LDSP + k] = (__bf16)bR[u].w;
    }
  };

  loadA(0); loadB(0);
  for (int k0 = 0; k0 < K; k0 += BK) {
    storeTiles();
    __syncthreads();
    int kn = k0 + BK;
    if (kn < K) { loadA(kn); loadB(kn); }  // prefetch next tile into registers

    // A fragment (ISA 16-bit A 16x32 layout): lane half = l/16,
    // elems 0..7 = K half*8.., elems 8..15 = K 16+half*8..
    const int m     = (wave << 4) + (lane & 15);
    const int halfA = (lane >> 4) << 3;    // 0 or 8
    bf16x8 alo = *(const bf16x8*)&sA[m * LDSP + halfA];
    bf16x8 ahi = *(const bf16x8*)&sA[m * LDSP + 16 + halfA];
    bf16x16 afrag = __builtin_shufflevector(alo, ahi,
        0,1,2,3,4,5,6,7,8,9,10,11,12,13,14,15);
    const int kb = (lane >> 4) << 3;       // 0 or 8 (ISA 16-bit B 32x16 layout)
#pragma unroll
    for (int j = 0; j < 4; ++j) {
      int n = (j << 4) + (lane & 15);
      bf16x8 blo = *(const bf16x8*)&sB[n * LDSP + kb];
      bf16x8 bhi = *(const bf16x8*)&sB[n * LDSP + kb + 16];
      bf16x16 bfrag = __builtin_shufflevector(blo, bhi,
          0,1,2,3,4,5,6,7,8,9,10,11,12,13,14,15);
      acc[j] = __builtin_amdgcn_wmma_f32_16x16x32_bf16(
          false, afrag, false, bfrag, (short)0, acc[j], false, false);
    }
    __syncthreads();
  }

  // D layout: lane = 16*(M>=8) + Ncol, VGPR r -> M = r + 8*(lane/16)
  const int nl    = lane & 15;
  const int mhalf = (lane >> 4) << 3;
#pragma unroll
  for (int j = 0; j < 4; ++j) {
    if (colBase + (j << 4) >= Nc) continue;        // wave-uniform guard (Nc % 16 == 0)
    int col = colBase + (j << 4) + nl;
    float bv = bias ? bias[col] : 0.f;
#pragma unroll
    for (int r = 0; r < 8; ++r) {
      int row = rowBase + (wave << 4) + mhalf + r; // always in range (M % 128 == 0)
      float v = acc[j][r] + bv;
      C[(size_t)row * Nc + col] = eluf(v, act);
    }
  }
}

// ================= graph LayerNorm (per-graph over all nodes&channels) + ELU ===========
__global__ __launch_bounds__(256) void graph_ln_elu_kernel(
    const float* __restrict__ X, float* __restrict__ Y,
    const float* __restrict__ g, const float* __restrict__ b, int act)
{
  const int gi = blockIdx.x;                 // one block per graph
  const float* x = X + (size_t)gi * (NPG * CD);
  float* y = Y + (size_t)gi * (NPG * CD);
  __shared__ float rs[256], rq[256];
  float s = 0.f, q = 0.f;
  for (int i = threadIdx.x; i < NPG * CD; i += 256) { float v = x[i]; s += v; q += v * v; }
  rs[threadIdx.x] = s; rq[threadIdx.x] = q;
  __syncthreads();
  for (int o = 128; o > 0; o >>= 1) {
    if (threadIdx.x < o) { rs[threadIdx.x] += rs[threadIdx.x + o]; rq[threadIdx.x] += rq[threadIdx.x + o]; }
    __syncthreads();
  }
  const float invN = 1.f / (float)(NPG * CD);
  float mean = rs[0] * invN;
  float var  = rq[0] * invN - mean * mean;
  float inv  = rsqrtf(var + 1e-5f);
  for (int i = threadIdx.x; i < NPG * CD; i += 256) {
    int c = i & (CD - 1);
    float v = (x[i] - mean) * inv * g[c] + b[c];
    y[i] = eluf(v, act);
  }
}

// ================= edge attention: logits + running segment max =================
__global__ __launch_bounds__(256) void attn_logits_kernel(
    const float* __restrict__ Q, const float* __restrict__ Kt, const float* __restrict__ Eh,
    const int* __restrict__ src, const int* __restrict__ dst,
    float* __restrict__ logits, unsigned* __restrict__ segMax)
{
  int t = blockIdx.x * 256 + threadIdx.x;          // over NE*NH
  if (t >= NE * NH) return;
  int e = t >> 2, h = t & 3;
  int s = src[e], d = dst[e];
  const float* qp = Q  + (size_t)d * HC + h * CD;
  const float* kp = Kt + (size_t)s * HC + h * CD;
  const float* ep = Eh + (size_t)e * HC + h * CD;
  float acc = 0.f;
#pragma unroll
  for (int c = 0; c < CD; c += 4) {
    float4 qv = *(const float4*)(qp + c);
    float4 kv = *(const float4*)(kp + c);
    float4 ev = *(const float4*)(ep + c);
    acc += qv.x * (kv.x + ev.x) + qv.y * (kv.y + ev.y)
         + qv.z * (kv.z + ev.z) + qv.w * (kv.w + ev.w);
  }
  float lg = acc * 0.125f;                         // co^-0.5, co=64
  logits[t] = lg;
  atomicMax(&segMax[d * NH + h], f2key(lg));
}

// ================= edge attention: exp & segment sum =================
__global__ __launch_bounds__(256) void attn_expsum_kernel(
    const int* __restrict__ dst, float* __restrict__ logits,
    const unsigned* __restrict__ segMax, float* __restrict__ segSum)
{
  int t = blockIdx.x * 256 + threadIdx.x;
  if (t >= NE * NH) return;
  int e = t >> 2, h = t & 3;
  int d = dst[e];
  float m = key2f(segMax[d * NH + h]);
  float ex = __expf(logits[t] - m);
  logits[t] = ex;
  atomicAdd(&segSum[d * NH + h], ex);
}

// ============ message aggregation (head-mean folded in, atomicAdd into skip buffer) ======
__global__ __launch_bounds__(256) void attn_msg_kernel(
    const float* __restrict__ V, const float* __restrict__ Eh,
    const int* __restrict__ src, const int* __restrict__ dst,
    const float* __restrict__ exl, const float* __restrict__ segSum,
    float* __restrict__ out)
{
  long t = (long)blockIdx.x * 256 + threadIdx.x;   // over NE*HC
  if (t >= (long)NE * HC) return;
  int e = (int)(t >> 8);
  int hc = (int)(t & 255);
  int h = hc >> 6;
  int s = src[e], d = dst[e];
  float w = exl[e * NH + h] / (segSum[d * NH + h] + 1e-16f);
  float val = (V[(size_t)s * HC + hc] + Eh[(size_t)e * HC + hc]) * w * 0.25f; // /H
  atomicAdd(&out[(size_t)d * CD + (hc & 63)], val);
}

// ================= pooling: agg = segment_sum(x[src], dst) =================
__global__ __launch_bounds__(256) void scatter_add_kernel(
    const float* __restrict__ X, const int* __restrict__ src, const int* __restrict__ dst,
    float* __restrict__ agg)
{
  long t = (long)blockIdx.x * 256 + threadIdx.x;   // over NE*CD
  if (t >= (long)NE * CD) return;
  int e = (int)(t >> 6), c = (int)(t & 63);
  atomicAdd(&agg[(size_t)dst[e] * CD + c], X[(size_t)src[e] * CD + c]);
}

__global__ __launch_bounds__(256) void pool_score_kernel(
    const float* __restrict__ agg, const float* __restrict__ x,
    const float* __restrict__ Wrel, const float* __restrict__ brel,
    const float* __restrict__ Wroot, float* __restrict__ s)
{
  int n = blockIdx.x * 256 + threadIdx.x;
  if (n >= NN) return;
  float a = 0.f;
#pragma unroll 8
  for (int c = 0; c < CD; ++c)
    a += agg[(size_t)n * CD + c] * Wrel[c] + x[(size_t)n * CD + c] * Wroot[c];
  s[n] = a + brel[0];
}

// softmax over graph + weighted node sum (one block of 64 per graph; NPG==CD==64)
__global__ __launch_bounds__(64) void pool_graph_kernel(
    const float* __restrict__ x, const float* __restrict__ s, float* __restrict__ gx)
{
  const int g = blockIdx.x;
  const int t = threadIdx.x;
  __shared__ float sc[NPG], red[NPG];
  float v = s[g * NPG + t];
  red[t] = v; __syncthreads();
  for (int o = 32; o > 0; o >>= 1) { if (t < o) red[t] = fmaxf(red[t], red[t + o]); __syncthreads(); }
  float mx = red[0]; __syncthreads();
  float ex = __expf(v - mx);
  sc[t] = ex; red[t] = ex; __syncthreads();
  for (int o = 32; o > 0; o >>= 1) { if (t < o) red[t] += red[t + o]; __syncthreads(); }
  float z = red[0] + 1e-16f;
  __syncthreads();
  float acc = 0.f;
  for (int n = 0; n < NPG; ++n)
    acc += x[((size_t)g * NPG + n) * CD + t] * (sc[n] / z);
  gx[(size_t)g * CD + t] = acc;
}

// ================= full-tensor LayerNorm (two-pass) =================
__global__ __launch_bounds__(256) void reduce_sums_kernel(
    const float* __restrict__ X, long n, float* __restrict__ acc2)
{
  __shared__ float rs[256], rq[256];
  float s = 0.f, q = 0.f;
  for (long i = (long)blockIdx.x * 256 + threadIdx.x; i < n; i += (long)gridDim.x * 256) {
    float v = X[i]; s += v; q += v * v;
  }
  rs[threadIdx.x] = s; rq[threadIdx.x] = q;
  __syncthreads();
  for (int o = 128; o > 0; o >>= 1) {
    if (threadIdx.x < o) { rs[threadIdx.x] += rs[threadIdx.x + o]; rq[threadIdx.x] += rq[threadIdx.x + o]; }
    __syncthreads();
  }
  if (threadIdx.x == 0) { atomicAdd(&acc2[0], rs[0]); atomicAdd(&acc2[1], rq[0]); }
}

__global__ __launch_bounds__(256) void full_ln_elu_kernel(
    const float* __restrict__ X, float* __restrict__ Y,
    const float* __restrict__ g, const float* __restrict__ b,
    const float* __restrict__ acc2, long n, int C)
{
  float mean = acc2[0] / (float)n;
  float var  = acc2[1] / (float)n - mean * mean;
  float inv  = rsqrtf(var + 1e-5f);
  for (long i = (long)blockIdx.x * 256 + threadIdx.x; i < n; i += (long)gridDim.x * 256) {
    int c = (int)(i % C);
    float v = (X[i] - mean) * inv * g[c] + b[c];
    Y[i] = eluf(v, 1);
  }
}

// ge = segment_sum(ea, ebatch): edges contiguous per graph
__global__ __launch_bounds__(64) void seg_sum_edges_kernel(
    const float* __restrict__ ea, float* __restrict__ ge)
{
  int g = blockIdx.x, c = threadIdx.x;     // C = 64
  float acc = 0.f;
  for (int i = 0; i < EPG; ++i)
    acc += ea[((size_t)g * EPG + i) * CD + c];
  ge[(size_t)g * CD + c] = acc;
}

__global__ __launch_bounds__(256) void mul_kernel(
    const float* __restrict__ a, const float* __restrict__ b, float* __restrict__ c, int n)
{
  int i = blockIdx.x * 256 + threadIdx.x;
  if (i < n) c[i] = a[i] * b[i];
}

// row L2-normalize, write with stride (into reprs[:, blk, :])
__global__ __launch_bounds__(128) void l2norm_rows_kernel(
    const float* __restrict__ X, float* __restrict__ Y, int outStride)
{
  int r = blockIdx.x, t = threadIdx.x;     // C = 128
  __shared__ float red[128];
  float v = X[(size_t)r * HIDD + t];
  red[t] = v * v; __syncthreads();
  for (int o = 64; o > 0; o >>= 1) { if (t < o) red[t] += red[t + o]; __syncthreads(); }
  float scale = 1.f / fmaxf(sqrtf(red[0]), 1e-12f);
  Y[(size_t)r * outStride + t] = v * scale;
}

// row LayerNorm, C=128
__global__ __launch_bounds__(128) void row_ln_kernel(
    const float* __restrict__ X, float* __restrict__ Y,
    const float* __restrict__ g, const float* __restrict__ b)
{
  int r = blockIdx.x, t = threadIdx.x;
  __shared__ float rs[128], rq[128];
  float v = X[(size_t)r * HIDD + t];
  rs[t] = v; rq[t] = v * v; __syncthreads();
  for (int o = 64; o > 0; o >>= 1) { if (t < o) { rs[t] += rs[t + o]; rq[t] += rq[t + o]; } __syncthreads(); }
  float mean = rs[0] * (1.f / HIDD);
  float var  = rq[0] * (1.f / HIDD) - mean * mean;
  Y[(size_t)r * HIDD + t] = (v - mean) * rsqrtf(var + 1e-5f) * g[t] + b[t];
}

// co-attention core: per (b, head, token): softmax over 2 keys, weighted V
__global__ __launch_bounds__(128) void coatt_attn_kernel(
    const float* __restrict__ q, const float* __restrict__ kv, float* __restrict__ o)
{
  const int idx = blockIdx.x;
  const int n = idx & 1, h = (idx >> 1) & 1, b = idx >> 2;
  const int t = threadIdx.x;
  const float* qp = q  + ((size_t)(b * 2 + n)) * 256 + h * 128;
  const float* k0 = kv + ((size_t)(b * 2 + 0)) * 512 + h * 128;
  const float* k1 = kv + ((size_t)(b * 2 + 1)) * 512 + h * 128;
  const float* v0 = k0 + 256;
  const float* v1 = k1 + 256;
  __shared__ float r0[128], r1[128];
  float qd = qp[t];
  r0[t] = qd * k0[t]; r1[t] = qd * k1[t];
  __syncthreads();
  for (int off = 64; off > 0; off >>= 1) {
    if (t < off) { r0[t] += r0[t + off]; r1[t] += r1[t + off]; }
    __syncthreads();
  }
  const float scl = 0.0883883476483184f;   // 128^-0.5
  float s0 = r0[0] * scl, s1 = r1[0] * scl;
  float mx = fmaxf(s0, s1);
  float e0 = __expf(s0 - mx), e1 = __expf(s1 - mx);
  float iz = 1.f / (e0 + e1);
  o[((size_t)(b * 2 + n)) * 256 + h * 128 + t] = (e0 * v0[t] + e1 * v1[t]) * iz;
}

// m[b,n,m'] = <hA[b,n,:], tA[b,m',:]>
__global__ __launch_bounds__(128) void pair_dot_kernel(
    const float* __restrict__ hA, const float* __restrict__ tA, float* __restrict__ mb)
{
  const int idx = blockIdx.x;
  const int mm = idx & 1, nn = (idx >> 1) & 1, b = idx >> 2;
  const int t = threadIdx.x;
  __shared__ float red[128];
  red[t] = hA[((size_t)(b * 2 + nn)) * HIDD + t] * tA[((size_t)(b * 2 + mm)) * HIDD + t];
  __syncthreads();
  for (int o = 64; o > 0; o >>= 1) { if (t < o) red[t] += red[t + o]; __syncthreads(); }
  if (t == 0) mb[b * 4 + nn * 2 + mm] = red[0];
}

// out = softmax(m @ Wm + bm) per batch row; m is [B,4], Wm [4,65]
__global__ __launch_bounds__(128) void final_softmax_kernel(
    const float* __restrict__ mb, const float* __restrict__ Wm,
    const float* __restrict__ bm, float* __restrict__ out)
{
  const int b = blockIdx.x, t = threadIdx.x;
  __shared__ float red[128];
  float m0 = mb[b * 4 + 0], m1 = mb[b * 4 + 1], m2 = mb[b * 4 + 2], m3 = mb[b * 4 + 3];
  float lv = -1e30f;
  if (t < NCLS)
    lv = m0 * Wm[t] + m1 * Wm[NCLS + t] + m2 * Wm[2 * NCLS + t] + m3 * Wm[3 * NCLS + t] + bm[t];
  red[t] = lv; __syncthreads();
  for (int o = 64; o > 0; o >>= 1) { if (t < o) red[t] = fmaxf(red[t], red[t + o]); __syncthreads(); }
  float mx = red[0]; __syncthreads();
  float ev = (t < NCLS) ? __expf(lv - mx) : 0.f;
  red[t] = ev; __syncthreads();
  for (int o = 64; o > 0; o >>= 1) { if (t < o) red[t] += red[t + o]; __syncthreads(); }
  float z = red[0];
  if (t < NCLS) out[(size_t)b * NCLS + t] = ev / z;
}

// =============================== host orchestration ===============================
extern "C" void kernel_launch(void* const* d_in, const int* in_sizes, int n_in,
                              void* d_out, int out_size, void* d_ws, size_t ws_size,
                              hipStream_t stream) {
  (void)in_sizes; (void)n_in; (void)out_size; (void)ws_size;
  auto P = [&](int i) -> const float* { return (const float*)d_in[i]; };

  // ---- inputs (setup_inputs dict order) ----
  const float* x1  = P(0);
  const int*   ei1 = (const int*)d_in[1];
  const float* ea1 = P(2);
  const float* x2  = P(5);
  const int*   ei2 = (const int*)d_in[6];
  const float* ea2 = P(7);

  // ---- params: jax pytree flatten (sorted dict keys, capitals first) ----
  // top: Wei(10) Win(11) Wm(12) bei(13) bin(14) blocks0(15..48) blocks1(49..82)
  //      bm(83) co{Wkv(84) Wout(85) Wq(86) b(87) g(88)} nb(89) ng(90)
  // block leaves: +0 Wre +1 Wrel +2 Wroot +3 Wtr +4 Wu1 +5 Wu2 +6 bre +7 brel
  //      +8 bu1 +9 bu2 +10..18 conv1 +19..27 conv2 +28 n1b +29 n1g +30 n2b +31 n2g +32 neb +33 neg
  // conv leaves: +0 We +1 Wk +2 Wq +3 Ws +4 Wv +5 bk +6 bq +7 bs +8 bv
  const float* Wei  = P(10); const float* Win = P(11); const float* Wm  = P(12);
  const float* bei  = P(13); const float* bin = P(14);
  const float* bm   = P(83);
  const float* coWkv = P(84); const float* coWout = P(85); const float* coWq = P(86);
  const float* cob  = P(87); const float* cog = P(88);
  const float* nb_  = P(89); const float* ng_ = P(90);

  // ---- workspace layout (floats) ----
  size_t o = 0;
  auto alloc = [&](size_t n) { float* p = (float*)d_ws + o; o += n; return p; };
  float* Qb     = alloc((size_t)NN * HC);
  float* Kb     = alloc((size_t)NN * HC);
  float* Vb     = alloc((size_t)NN * HC);
  float* Ehb    = alloc((size_t)NE * HC);
  float* xA     = alloc((size_t)NN * CD);
  float* xB     = alloc((size_t)NN * CD);
  float* eaA    = alloc((size_t)NE * CD);
  float* eaB    = alloc((size_t)NE * CD);
  float* logits = alloc((size_t)NE * NH);
  unsigned* segMax = (unsigned*)alloc((size_t)NN * NH);
  float* segSum = alloc((size_t)NN * NH);
  float* aggX   = alloc((size_t)NN * CD);
  float* sScore = alloc(NN);
  float* gx     = alloc((size_t)NGR * CD);
  float* ge     = alloc((size_t)NGR * CD);
  float* gxge   = alloc((size_t)NGR * CD);
  float* gtmp   = alloc((size_t)NGR * HIDD);
  float* reprH  = alloc((size_t)NGR * 2 * HIDD);
  float* reprT  = alloc((size_t)NGR * 2 * HIDD);
  float* acc2   = alloc(8);
  float* qn     = alloc((size_t)NGR * 2 * HIDD);
  float* kvn    = alloc((size_t)NGR * 2 * HIDD);
  float* qbuf   = alloc((size_t)NGR * 2 * 256);
  float* kvbuf  = alloc((size_t)NGR * 2 * 512);
  float* attout = alloc((size_t)NGR * 2 * 256);
  float* hAb    = alloc((size_t)NGR * 2 * HIDD);
  float* tAb    = alloc((size_t)NGR * 2 * HIDD);
  float* mbuf   = alloc((size_t)NGR * 4);

  auto gemm = [&](const float* A, const float* W, const float* bias, float* C,
                  int M, int K, int Nc, int act) {
    dim3 grid((Nc + BN - 1) / BN, (M + BM - 1) / BM);
    gemm_wmma_kernel<<<grid, 256, 0, stream>>>(A, W, bias, C, M, K, Nc, act);
  };

  const int gEH = (NE * NH + 255) / 256;
  auto runConv = [&](const float* xin, float* xout, const float* eaIn, int eDim,
                     int cb, const float* lng, const float* lnb,
                     const int* src, const int* dst) {
    gemm(xin, P(cb + 2), P(cb + 6), Qb, NN, CD, HC, 0);    // Q = x@Wq+bq
    gemm(xin, P(cb + 1), P(cb + 5), Kb, NN, CD, HC, 0);    // K = x@Wk+bk
    gemm(xin, P(cb + 4), P(cb + 8), Vb, NN, CD, HC, 0);    // V = x@Wv+bv
    gemm(eaIn, P(cb + 0), nullptr, Ehb, NE, eDim, HC, 0);  // Eh = ea@We
    hipMemsetAsync(segMax, 0, (size_t)NN * NH * sizeof(unsigned), stream);
    hipMemsetAsync(segSum, 0, (size_t)NN * NH * sizeof(float), stream);
    attn_logits_kernel<<<gEH, 256, 0, stream>>>(Qb, Kb, Ehb, src, dst, logits, segMax);
    attn_expsum_kernel<<<gEH, 256, 0, stream>>>(dst, logits, segMax, segSum);
    gemm(xin, P(cb + 3), P(cb + 7), xout, NN, CD, CD, 0);  // skip: x@Ws+bs
    attn_msg_kernel<<<(int)(((long)NE * HC + 255) / 256), 256, 0, stream>>>(
        Vb, Ehb, src, dst, logits, segSum, xout);
    graph_ln_elu_kernel<<<NGR, 256, 0, stream>>>(xout, xout, lng, lnb, 1);
  };

  auto runDrug = [&](const float* xin, const int* ei, const float* eain, float* repr) {
    const int* src = ei;
    const int* dst = ei + NE;
    gemm(xin, Win, bin, xA, NN, CD, CD, 0);
    graph_ln_elu_kernel<<<NGR, 256, 0, stream>>>(xA, xA, ng_, nb_, 1);
    gemm(eain, Wei, bei, eaA, NE, 16, 16, 1);
    float* x = xA; float* xo = xB;
    float* ea = eaA; float* eo = eaB;
    for (int blk = 0; blk < 2; ++blk) {
      const int bb = 15 + blk * 34;
      const int eDim = (blk == 0) ? 16 : 64;
      runConv(x, xo, ea, eDim, bb + 10, P(bb + 29), P(bb + 28), src, dst);
      { float* t = x; x = xo; xo = t; }
      gemm(ea, P(bb + 4), P(bb + 8), eo, NE, eDim, eDim, 1);   // ea = elu(ea@Wu1+bu1)
      { float* t = ea; ea = eo; eo = t; }
      runConv(x, xo, ea, eDim, bb + 19, P(bb + 31), P(bb + 30), src, dst);
      { float* t = x; x = xo; xo = t; }
      gemm(ea, P(bb + 5), P(bb + 9), eo, NE, eDim, eDim, 1);   // ea = elu(ea@Wu2+bu2)
      { float* t = ea; ea = eo; eo = t; }
      // SAG pooling (min_score=-1): scores + graph softmax reweight
      hipMemsetAsync(aggX, 0, (size_t)NN * CD * sizeof(float), stream);
      scatter_add_kernel<<<(int)(((long)NE * CD + 255) / 256), 256, 0, stream>>>(x, src, dst, aggX);
      pool_score_kernel<<<(NN + 255) / 256, 256, 0, stream>>>(
          aggX, x, P(bb + 1), P(bb + 7), P(bb + 2), sScore);
      pool_graph_kernel<<<NGR, 64, 0, stream>>>(x, sScore, gx);
      // ea -> e2 dims, full LN, elu
      gemm(ea, P(bb + 0), P(bb + 6), eo, NE, eDim, CD, 0);     // ea@Wre+bre -> [NE,64]
      hipMemsetAsync(acc2, 0, 2 * sizeof(float), stream);
      reduce_sums_kernel<<<1024, 256, 0, stream>>>(eo, (long)NE * CD, acc2);
      full_ln_elu_kernel<<<1024, 256, 0, stream>>>(eo, ea, P(bb + 33), P(bb + 32),
                                                   acc2, (long)NE * CD, CD);
      seg_sum_edges_kernel<<<NGR, 64, 0, stream>>>(ea, ge);
      mul_kernel<<<(NGR * CD + 255) / 256, 256, 0, stream>>>(gx, ge, gxge, NGR * CD);
      gemm(gxge, P(bb + 3), nullptr, gtmp, NGR, CD, HIDD, 0); // (gx*ge)@Wtr
      l2norm_rows_kernel<<<NGR, 128, 0, stream>>>(gtmp, repr + blk * HIDD, 2 * HIDD);
    }
  };

  runDrug(x1, ei1, ea1, reprH);
  runDrug(x2, ei2, ea2, reprT);

  // ---- co-attention (both directions) ----
  auto coatt = [&](const float* qin, const float* kvin, float* outp) {
    row_ln_kernel<<<NGR * 2, 128, 0, stream>>>(qin, qn, cog, cob);
    row_ln_kernel<<<NGR * 2, 128, 0, stream>>>(kvin, kvn, cog, cob);
    gemm(qn, coWq, nullptr, qbuf, NGR * 2, HIDD, 256, 0);
    gemm(kvn, coWkv, nullptr, kvbuf, NGR * 2, HIDD, 512, 0);
    coatt_attn_kernel<<<NGR * 4, 128, 0, stream>>>(qbuf, kvbuf, attout);
    gemm(attout, coWout, nullptr, outp, NGR * 2, 256, HIDD, 0);
  };
  coatt(reprH, reprT, hAb);
  coatt(reprT, reprH, tAb);

  pair_dot_kernel<<<NGR * 4, 128, 0, stream>>>(hAb, tAb, mbuf);
  final_softmax_kernel<<<NGR, 128, 0, stream>>>(mbuf, Wm, bm, (float*)d_out);
}
